// InfoVAELoss_58566174048494
// MI455X (gfx1250) — matrix operand order repaired and checked by reading, not compiled, not run on hardware
//
#include <hip/hip_runtime.h>

typedef __attribute__((ext_vector_type(16))) __bf16 v16bf;
typedef __attribute__((ext_vector_type(8)))  float  v8f;

typedef unsigned short u16;
typedef unsigned int   u32;

__device__ __forceinline__ u16 f32_to_bf16_rne(float f) {
    u32 u = __builtin_bit_cast(u32, f);
    u32 r = u + 0x7FFFu + ((u >> 16) & 1u);   // round-to-nearest-even (randn inputs: no NaN/Inf)
    return (u16)(r >> 16);
}
__device__ __forceinline__ float bf16_to_f32(u16 h) {
    u32 u = ((u32)h) << 16;
    return __builtin_bit_cast(float, u);
}

union Frag {
    uint4 u[2];
    v16bf v;
};

// B tile row stride in u16: 128 data + 8 pad = 272B rows.
//  - keeps every 16B ds_load_b128 aligned (272 = 17*16)
//  - row-to-row bank stride = 68 dwords ≡ 4 banks (mod 64) -> lanes spread over banks
#define BROW_U16 136

// ---------------------------------------------------------------------------
// Prep: split each f32 row into bf16 hi/lo and compute exact f32 row norms.
// One 128-thread block per (padded) row. Pad rows are zero-filled.
// ---------------------------------------------------------------------------
__global__ __launch_bounds__(128)
void prep_kernel(const float* __restrict__ X, int validRows,
                 u16* __restrict__ H, u16* __restrict__ L,
                 float* __restrict__ Nrm) {
    const int r = blockIdx.x;      // padded row index
    const int d = threadIdx.x;     // 0..127 (latent dim)
    const float x = (r < validRows) ? X[(size_t)r * 128 + d] : 0.0f;
    const u16 hi = f32_to_bf16_rne(x);
    const u16 lo = f32_to_bf16_rne(x - bf16_to_f32(hi));
    H[(size_t)r * 128 + d] = hi;
    L[(size_t)r * 128 + d] = lo;

    __shared__ float red[128];
    red[d] = x * x;
    __syncthreads();
    for (int s = 64; s > 0; s >>= 1) {
        if (d < s) red[d] += red[d + s];
        __syncthreads();
    }
    if (d == 0) Nrm[r] = red[0];
}

// ---------------------------------------------------------------------------
// Main: sum of exp(-(x2[m] + y2[n] - 2*dot(x_m, y_n)) / 16384) over a tile
// grid.  8 wave32s per block; the 16-column B tile (hi+lo bf16) is staged to
// LDS once per block via GLOBAL_LOAD_ASYNC_TO_LDS_B128 (ASYNCcnt), then each
// wave computes one 16x16 Gram tile with compensated bf16 WMMA
// (hi*hi + hi*lo + lo*hi), applies exp, and the block writes one partial sum
// (deterministic tree reductions only).
// ---------------------------------------------------------------------------
__global__ __launch_bounds__(256)
void gram_exp_kernel(const u16* __restrict__ Xh, const u16* __restrict__ Xl,
                     const float* __restrict__ Xn, int Mrows, int MrowTiles,
                     const u16* __restrict__ Yh, const u16* __restrict__ Yl,
                     const float* __restrict__ Yn,
                     float* __restrict__ partials) {
    const int lane = threadIdx.x & 31;
    const int wave = threadIdx.x >> 5;
    const int half = lane >> 4;     // lane group (0: lanes 0-15, 1: lanes 16-31)
    const int l16  = lane & 15;
    const int rowTile = blockIdx.y * 8 + wave;
    const int n0 = blockIdx.x * 16;

    // --- async-stage the shared B tile: [hi|lo] x 16 rows x 128 u16 --------
    __shared__ u16 btile[2 * 16 * BROW_U16];   // 8704 B
    {
        // 512 16-byte chunks: [mat(2)][row(16)][chunk(16)], 2 per thread
        #pragma unroll
        for (int c = threadIdx.x; c < 512; c += 256) {
            const int mat = c >> 8;          // 0 = hi, 1 = lo
            const int row = (c >> 4) & 15;
            const int ch  = c & 15;          // 16B chunk within row
            const u16* gsrc = (mat ? Yl : Yh) + (size_t)(n0 + row) * 128 + ch * 8;
            const u32 loff =
                (u32)(size_t)&btile[(mat * 16 + row) * BROW_U16 + ch * 8];
            asm volatile("global_load_async_to_lds_b128 %0, %1, off"
                         :: "v"(loff), "v"(gsrc) : "memory");
        }
        asm volatile("s_wait_asynccnt 0x0" ::: "memory");
    }
    __syncthreads();

    float lsum = 0.0f;
    if (rowTile < MrowTiles) {      // wave-uniform: EXEC stays all-ones for WMMA
        const int m0 = rowTile * 16;
        const int am = m0 + l16;    // A-fragment source row (X)
        const int bn = n0 + l16;    // B-fragment source column = Y row

        // per-lane LDS row pointers for B fragments
        const u16* bh_row = &btile[l16 * BROW_U16];
        const u16* bl_row = &btile[(16 + l16) * BROW_U16];

        v8f acc = {};
        #pragma unroll
        for (int kc = 0; kc < 128; kc += 32) {
            // A layout (16-bit, 16x32): lane half h holds K = kc+8h+0..7 and kc+16+8h+0..7
            const u16* pah = Xh + (size_t)am * 128 + kc + 8 * half;
            const u16* pal = Xl + (size_t)am * 128 + kc + 8 * half;
            Frag ah, al;
            ah.u[0] = *(const uint4*)(pah);
            ah.u[1] = *(const uint4*)(pah + 16);
            al.u[0] = *(const uint4*)(pal);
            al.u[1] = *(const uint4*)(pal + 16);
            // B layout (16-bit, 32x16): lane half h holds K = kc+8h+0..7 and kc+16+8h+0..7 (from LDS)
            const u16* pbh = bh_row + kc + 8 * half;
            const u16* pbl = bl_row + kc + 8 * half;
            Frag bh, bl;
            bh.u[0] = *(const uint4*)(pbh);
            bh.u[1] = *(const uint4*)(pbh + 16);
            bl.u[0] = *(const uint4*)(pbl);
            bl.u[1] = *(const uint4*)(pbl + 16);

            // Compensated product: hi*hi + hi*lo + lo*hi (lo*lo ~ 2^-16 rel, dropped)
            acc = __builtin_amdgcn_wmma_f32_16x16x32_bf16(false, ah.v, false, bh.v,
                                                          (short)0, acc, false, false);
            acc = __builtin_amdgcn_wmma_f32_16x16x32_bf16(false, ah.v, false, bl.v,
                                                          (short)0, acc, false, false);
            acc = __builtin_amdgcn_wmma_f32_16x16x32_bf16(false, al.v, false, bh.v,
                                                          (short)0, acc, false, false);
        }

        // C/D layout: VGPR r -> M = m0 + r + 8*half, N = n0 + (lane&15)
        const float yn = Yn[bn];
        #pragma unroll
        for (int r = 0; r < 8; ++r) {
            const int m = m0 + r + 8 * half;
            if (m < Mrows) {
                const float sq = Xn[m] + yn - 2.0f * acc[r];
                lsum += __expf(-sq * (1.0f / 16384.0f));
            }
        }
    }

    // wave32 reduction
    #pragma unroll
    for (int off = 16; off > 0; off >>= 1)
        lsum += __shfl_xor(lsum, off, 32);

    __shared__ float wred[8];
    if (lane == 0) wred[wave] = lsum;
    __syncthreads();
    if (threadIdx.x == 0) {
        float s = 0.0f;
        #pragma unroll
        for (int w = 0; w < 8; ++w) s += wred[w];
        partials[(size_t)blockIdx.y * gridDim.x + blockIdx.x] = s;
    }
}

// ---------------------------------------------------------------------------
// Finalize: deterministic reduction of the two partial arrays ->
//   out = S_zz / N^2 - S_tz / (M*N)
// ---------------------------------------------------------------------------
__global__ __launch_bounds__(256)
void finalize_kernel(const float* __restrict__ pzz, int nzz,
                     const float* __restrict__ ptz, int ntz,
                     float invZZ, float invTZ, float* __restrict__ out) {
    __shared__ float s1[256];
    __shared__ float s2[256];
    float a = 0.0f, b = 0.0f;
    for (int i = threadIdx.x; i < nzz; i += 256) a += pzz[i];
    for (int i = threadIdx.x; i < ntz; i += 256) b += ptz[i];
    s1[threadIdx.x] = a;
    s2[threadIdx.x] = b;
    __syncthreads();
    for (int s = 128; s > 0; s >>= 1) {
        if (threadIdx.x < s) {
            s1[threadIdx.x] += s1[threadIdx.x + s];
            s2[threadIdx.x] += s2[threadIdx.x + s];
        }
        __syncthreads();
    }
    if (threadIdx.x == 0) out[0] = s1[0] * invZZ - s2[0] * invTZ;
}

// ---------------------------------------------------------------------------
extern "C" void kernel_launch(void* const* d_in, const int* in_sizes, int n_in,
                              void* d_out, int out_size, void* d_ws, size_t ws_size,
                              hipStream_t stream) {
    const float* z  = (const float*)d_in[0];   // [N, 128]
    const float* zt = (const float*)d_in[1];   // [M, 128]
    const int D = 128;
    const int N    = in_sizes[0] / D;                  // 8192
    const int M    = in_sizes[1] / D;                  // 1000
    const int Mpad = ((M + 15) / 16) * 16;             // 1008

    // Workspace carve-up (256B aligned sections)
    char* ws = (char*)d_ws;
    size_t off = 0;
    auto carve = [&](size_t bytes) -> void* {
        void* p = ws + off;
        off = (off + bytes + 255) & ~(size_t)255;
        return p;
    };
    u16*   zh = (u16*)  carve((size_t)N * D * sizeof(u16));
    u16*   zl = (u16*)  carve((size_t)N * D * sizeof(u16));
    u16*   th = (u16*)  carve((size_t)Mpad * D * sizeof(u16));
    u16*   tl = (u16*)  carve((size_t)Mpad * D * sizeof(u16));
    float* zn = (float*)carve((size_t)N * sizeof(float));
    float* tn = (float*)carve((size_t)Mpad * sizeof(float));

    const int colTiles = N / 16;                       // 512
    const int zzTiles  = N / 16;                       // 512 row tiles
    const int tzTiles  = Mpad / 16;                    // 63 row tiles
    const int zzGroups = (zzTiles + 7) / 8;            // 64
    const int tzGroups = (tzTiles + 7) / 8;            // 8
    const int nzz = colTiles * zzGroups;
    const int ntz = colTiles * tzGroups;
    float* pzz = (float*)carve((size_t)nzz * sizeof(float));
    float* ptz = (float*)carve((size_t)ntz * sizeof(float));

    // 1) bf16 hi/lo split + row norms
    prep_kernel<<<N,    128, 0, stream>>>(z,  N, zh, zl, zn);
    prep_kernel<<<Mpad, 128, 0, stream>>>(zt, M, th, tl, tn);

    // 2) Gram + exp partial sums: K(z,z) and K(z_true, z)
    gram_exp_kernel<<<dim3(colTiles, zzGroups), 256, 0, stream>>>(
        zh, zl, zn, N, zzTiles, zh, zl, zn, pzz);
    gram_exp_kernel<<<dim3(colTiles, tzGroups), 256, 0, stream>>>(
        th, tl, tn, M, tzTiles, zh, zl, zn, ptz);

    // 3) scalar: mean(K_zz) - mean(K_tz)
    const float invZZ = 1.0f / ((float)N * (float)N);
    const float invTZ = 1.0f / ((float)M * (float)N);
    finalize_kernel<<<1, 256, 0, stream>>>(pzz, nzz, ptz, ntz, invZZ, invTZ,
                                           (float*)d_out);
}
